// CausalAttention_40312563040529
// MI455X (gfx1250) — matrix operand order repaired and checked by reading, not compiled, not run on hardware
//
#include <hip/hip_runtime.h>
#include <cstdint>
#include <cstddef>

// Problem constants (match reference)
#define BB 4
#define LL 4096
#define DD 2048
#define HH 128   // head dim

typedef _Float16 h16;
typedef __attribute__((ext_vector_type(16))) _Float16 v16h;
typedef __attribute__((ext_vector_type(8)))  float    v8f;
typedef __attribute__((ext_vector_type(4)))  unsigned int v4u;
typedef __attribute__((ext_vector_type(8)))  int      v8i;
typedef __attribute__((ext_vector_type(4)))  int      v4i;

#if defined(__has_builtin)
#  if __has_builtin(__builtin_amdgcn_tensor_load_to_lds)
#    define CA_HAVE_TDM 1
#  endif
#endif
#ifndef CA_HAVE_TDM
#  define CA_HAVE_TDM 0
#endif

union FragU { uint4 u[2]; v16h v; };

// A/B fragment = 16 halves per lane as two 16B loads (global or LDS).
__device__ __forceinline__ v16h load_frag(const h16* p0, const h16* p1) {
  FragU f;
  f.u[0] = *reinterpret_cast<const uint4*>(p0);
  f.u[1] = *reinterpret_cast<const uint4*>(p1);
  return f.v;
}

__device__ __forceinline__ v8f wmma16(v16h a, v16h b, v8f c) {
  return __builtin_amdgcn_wmma_f32_16x16x32_f16(
      false, a, false, b, (short)0, c, false, false);
}

__device__ __forceinline__ void wave_lds_fence() {
  asm volatile("s_wait_dscnt 0x0" ::: "memory");
}

__device__ __forceinline__ void tdm_wait0() {
#if CA_HAVE_TDM
#  if __has_builtin(__builtin_amdgcn_s_wait_tensorcnt)
  __builtin_amdgcn_s_wait_tensorcnt(0);
#  else
  asm volatile("s_wait_tensorcnt 0x0" ::: "memory");
#  endif
#endif
}

// Issue a TDM 2D tile load: rows n = 0..127 of BT starting at (n0, kelem),
// 32 f16 per row, row stride ldb elements -> LDS [128][32] contiguous.
__device__ __forceinline__ void tdm_load_b_panel(const h16* BT, int ldb, int n0,
                                                 int kelem, void* lds_dst) {
#if CA_HAVE_TDM
  const unsigned long long ga =
      (unsigned long long)(uintptr_t)(BT + (size_t)n0 * ldb + kelem);
  const unsigned la = (unsigned)(size_t)lds_dst;  // low 32 bits = LDS offset

  v4u g0;
  g0.x = 1u;                                   // count=1, no gather/restore
  g0.y = la;                                   // lds_addr
  g0.z = (unsigned)ga;                         // global_addr[31:0]
  g0.w = (unsigned)(ga >> 32) | (2u << 30);    // global_addr[56:32] | type=2

  const unsigned td0 = 32u, td1 = 128u, tl0 = 32u, tl1 = 128u;
  const unsigned st0 = (unsigned)ldb;
  v8i g1;
  g1[0] = (int)(1u << 16);                     // data_size=1 (2 bytes)
  g1[1] = (int)((td0 & 0xFFFFu) << 16);        // tensor_dim0[15:0]
  g1[2] = (int)(((td0 >> 16) & 0xFFFFu) | ((td1 & 0xFFFFu) << 16));
  g1[3] = (int)(((td1 >> 16) & 0xFFFFu) | (tl0 << 16));   // tile_dim0
  g1[4] = (int)tl1;                            // tile_dim1 (tile_dim2=0)
  g1[5] = (int)st0;                            // tensor_dim0_stride[31:0]
  g1[6] = 0;
  g1[7] = 0;

  v4i gz = {0, 0, 0, 0};
#  if defined(__clang_major__) && (__clang_major__ >= 23)
  v8i gz8 = {0, 0, 0, 0, 0, 0, 0, 0};
  __builtin_amdgcn_tensor_load_to_lds(g0, g1, gz, gz, gz8, 0);
#  else
  __builtin_amdgcn_tensor_load_to_lds(g0, g1, gz, gz, 0);
#  endif
#else
  (void)BT; (void)ldb; (void)n0; (void)kelem; (void)lds_dst;
#endif
}

// ---------------- conversion kernels ----------------

__global__ void ca_cvt_f32_to_f16(const float* __restrict__ in,
                                  h16* __restrict__ out, int n) {
  int i = blockIdx.x * blockDim.x + threadIdx.x;
  int stride = gridDim.x * blockDim.x;
  for (; i < n; i += stride) out[i] = (h16)in[i];
}

__global__ void ca_tcvt_f32_to_f16(const float* __restrict__ in,
                                   h16* __restrict__ out, int R, int C) {
  int i = blockIdx.x * blockDim.x + threadIdx.x;
  int stride = gridDim.x * blockDim.x;
  int total = R * C;
  for (; i < total; i += stride) {
    int r = i / C, c = i % C;
    out[(size_t)c * R + r] = (h16)in[i];
  }
}

// ---------------- generic GEMM: C = A @ BT^T (+bias) ----------------
// B panel (block-uniform, 128 cols x 32 k = 8KB) staged into LDS by the
// Tensor Data Mover, double buffered; fragments read via ds_load_b128.
__global__ __launch_bounds__(128)
void ca_gemm(const h16* __restrict__ A, const h16* __restrict__ BT,
             const float* __restrict__ bias, void* __restrict__ Cout,
             int M, int N, int K, int lda, int ldb, int ldc, int mode) {
  __shared__ h16 sB[2][128][32];

  const int tid   = threadIdx.x;
  const int w     = tid >> 5;
  const int lane  = tid & 31;
  const int col   = lane & 15;
  const int half8 = (lane >> 4) * 8;
  const int hb16  = (lane >> 4) * 16;
  const int m0 = (blockIdx.y * 4 + w) * 16;
  const int n0 = blockIdx.x * 128;

#if !CA_HAVE_TDM
  auto stage = [&](int kelem, int nb) {
    const uint4* s4 = (const uint4*)(BT + (size_t)(n0 + tid) * ldb + kelem);
    uint4* d4 = (uint4*)&sB[nb][tid][0];
    d4[0] = s4[0]; d4[1] = s4[1]; d4[2] = s4[2]; d4[3] = s4[3];
  };
#endif

#if CA_HAVE_TDM
  if (tid < 32) {
    tdm_load_b_panel(BT, ldb, n0, 0, &sB[0][0][0]);
    tdm_wait0();
  }
#else
  stage(0, 0);
#endif
  __syncthreads();

  const h16* arow = A + (size_t)(m0 + col) * lda;

  v8f z = {0.f, 0.f, 0.f, 0.f, 0.f, 0.f, 0.f, 0.f};
  v8f acc[8];
#pragma unroll
  for (int nt = 0; nt < 8; ++nt) acc[nt] = z;

  v16h af = load_frag(arow + half8, arow + 16 + half8);
  int nb = 0;
  for (int kc = 0; kc < K; kc += 32) {
    const bool hasnext = (kc + 32 < K);
    if (hasnext) {
#if CA_HAVE_TDM
      if (tid < 32) tdm_load_b_panel(BT, ldb, n0, kc + 32, &sB[nb ^ 1][0][0]);
#else
      stage(kc + 32, nb ^ 1);
#endif
    }
    v16h afn = af;
    if (hasnext)
      afn = load_frag(arow + kc + 32 + half8, arow + kc + 48 + half8);

    v16h bf = load_frag(&sB[nb][col][half8], &sB[nb][col][16 + half8]);
#pragma unroll
    for (int nt = 0; nt < 8; ++nt) {
      v16h bn = bf;
      if (nt < 7)
        bn = load_frag(&sB[nb][(nt + 1) * 16 + col][half8],
                       &sB[nb][(nt + 1) * 16 + col][16 + half8]);
      acc[nt] = wmma16(af, bf, acc[nt]);
      bf = bn;
    }
    af = afn;

    if (hasnext) {
#if CA_HAVE_TDM
      if (tid < 32) tdm_wait0();
#endif
      __syncthreads();
      nb ^= 1;
    }
  }

#pragma unroll
  for (int nt = 0; nt < 8; ++nt) {
    const int n = n0 + nt * 16 + col;
    const float bv = bias ? bias[n] : 0.f;
#pragma unroll
    for (int r = 0; r < 8; ++r) {
      const int m = m0 + r + half8;
      const float o = acc[nt][r] + bv;
      if (mode == 0)      ((h16*)Cout)[(size_t)m * ldc + n] = (h16)o;
      else if (mode == 1) ((h16*)Cout)[(size_t)n * ldc + m] = (h16)o;
      else                ((float*)Cout)[(size_t)m * ldc + n] = o;
    }
  }
}

// ---------------- flash attention (causal, single head) ----------------
// Softmax row statistics via xor-shuffle reductions inside each 16-lane
// half-group (C-layout rows span lanes {0..15} or {16..31}); only the
// P tile goes through LDS for the C-layout -> A-layout transpose.
__global__ __launch_bounds__(128)
void ca_flash_attn(const h16* __restrict__ Q, const h16* __restrict__ Kb,
                   const h16* __restrict__ VT, h16* __restrict__ O) {
  __shared__ h16 sP[4][16][32];

  const int w     = threadIdx.x >> 5;
  const int lane  = threadIdx.x & 31;
  const int col   = lane & 15;
  const int half8 = (lane >> 4) * 8;
  const int hb16  = (lane >> 4) * 16;

  const int wg = blockIdx.x * 4 + w;
  const int b  = wg >> 8;            // L/16 = 256 tiles per batch
  const int q0 = (wg & 255) * 16;
  const float scale = 0.08838834764831845f;  // 1/sqrt(128)

  const h16* qrow = Q + (size_t)(b * LL + q0 + col) * HH;
  v16h qf[4];
#pragma unroll
  for (int kc = 0; kc < 4; ++kc)
    qf[kc] = load_frag(qrow + kc * 32 + half8, qrow + kc * 32 + 16 + half8);

  v8f z = {0.f, 0.f, 0.f, 0.f, 0.f, 0.f, 0.f, 0.f};
  v8f Oacc[8];
#pragma unroll
  for (int nt = 0; nt < 8; ++nt) Oacc[nt] = z;

  // Per-lane row statistics, replicated across the 16 lanes of a half-group.
  float rowmax[8], rowsum[8];
#pragma unroll
  for (int r = 0; r < 8; ++r) { rowmax[r] = -1e30f; rowsum[r] = 0.f; }

  const int kend = q0 + 16;
  for (int kbk = 0; kbk < kend; kbk += 32) {
    if (kbk + 32 < kend)
      __builtin_prefetch(Kb + (size_t)(b * LL + kbk + 32 + lane) * HH, 0, 0);

    // ---- S = Q @ K^T for 32 keys, with K-fragment lookahead ----
    const h16* k0p = Kb + (size_t)(b * LL + kbk + col) * HH + half8;
    const h16* k1p = k0p + (size_t)16 * HH;
    v16h kb0 = load_frag(k0p, k0p + 16);
    v16h kb1 = load_frag(k1p, k1p + 16);
    v8f S0 = z, S1 = z;
#pragma unroll
    for (int kc = 0; kc < 4; ++kc) {
      v16h n0f = kb0, n1f = kb1;
      if (kc < 3) {
        n0f = load_frag(k0p + (kc + 1) * 32, k0p + (kc + 1) * 32 + 16);
        n1f = load_frag(k1p + (kc + 1) * 32, k1p + (kc + 1) * 32 + 16);
      }
      S0 = wmma16(qf[kc], kb0, S0);
      S1 = wmma16(qf[kc], kb1, S1);
      kb0 = n0f; kb1 = n1f;
    }

    // ---- scale, mask, online softmax fully in registers ----
#pragma unroll
    for (int r = 0; r < 8; ++r) {
      const int m = r + half8;
      const int qidx = q0 + m;
      float v0 = S0[r] * scale;
      if (kbk + col > qidx) v0 = -1e30f;
      float v1 = S1[r] * scale;
      if (kbk + 16 + col > qidx) v1 = -1e30f;

      // row max over the 16-lane half-group (bit 4 never flips)
      float bm = fmaxf(v0, v1);
      bm = fmaxf(bm, __shfl_xor(bm, 1, 32));
      bm = fmaxf(bm, __shfl_xor(bm, 2, 32));
      bm = fmaxf(bm, __shfl_xor(bm, 4, 32));
      bm = fmaxf(bm, __shfl_xor(bm, 8, 32));

      const float mx = fmaxf(rowmax[r], bm);
      const float al = __expf(rowmax[r] - mx);
      const float e0 = __expf(v0 - mx);
      const float e1 = __expf(v1 - mx);

      float ps = e0 + e1;
      ps += __shfl_xor(ps, 1, 32);
      ps += __shfl_xor(ps, 2, 32);
      ps += __shfl_xor(ps, 4, 32);
      ps += __shfl_xor(ps, 8, 32);

      rowsum[r] = rowsum[r] * al + ps;
      rowmax[r] = mx;

      sP[w][m][col]      = (h16)e0;
      sP[w][m][16 + col] = (h16)e1;

#pragma unroll
      for (int nt = 0; nt < 8; ++nt) Oacc[nt][r] *= al;
    }
    wave_lds_fence();

    // ---- O += P(16x32) @ V(32x128), with V-fragment lookahead ----
    v16h pf = load_frag(&sP[w][col][half8], &sP[w][col][16 + half8]);
    const size_t vstride = (size_t)16 * (BB * LL);
    const h16* vrow = VT + (size_t)col * (BB * LL) + (size_t)b * LL + kbk + half8;
    v16h vf = load_frag(vrow, vrow + 16);
#pragma unroll
    for (int nt = 0; nt < 8; ++nt) {
      v16h vn = vf;
      if (nt < 7) {
        const h16* vr2 = vrow + (size_t)(nt + 1) * vstride;
        vn = load_frag(vr2, vr2 + 16);
      }
      Oacc[nt] = wmma16(pf, vf, Oacc[nt]);
      vf = vn;
    }
  }

  // ---- normalize and store (row sums already live in registers) ----
#pragma unroll
  for (int nt = 0; nt < 8; ++nt)
#pragma unroll
    for (int r = 0; r < 8; ++r) {
      const int m = r + half8;
      const int n = nt * 16 + col;
      O[(size_t)(b * LL + q0 + m) * HH + n] = (h16)(Oacc[nt][r] / rowsum[r]);
    }
}

// ---------------- launch ----------------

extern "C" void kernel_launch(void* const* d_in, const int* in_sizes, int n_in,
                              void* d_out, int out_size, void* d_ws, size_t ws_size,
                              hipStream_t stream) {
  const float* x  = (const float*)d_in[0];
  const float* Wq = (const float*)d_in[1];
  const float* bq = (const float*)d_in[2];
  const float* Wk = (const float*)d_in[3];
  const float* bk = (const float*)d_in[4];
  const float* Wv = (const float*)d_in[5];
  const float* bv = (const float*)d_in[6];
  const float* Wo = (const float*)d_in[7];
  const float* bo = (const float*)d_in[8];

  char* ws = (char*)d_ws;
  size_t off = 0;
  auto alloc = [&](size_t bytes) -> void* {
    void* p = ws + off;
    off += (bytes + 255) & ~(size_t)255;
    return p;
  };

  const size_t ML = (size_t)BB * LL;  // 16384 rows
  h16* xb  = (h16*)alloc(ML * DD * sizeof(h16));
  h16* wqT = (h16*)alloc((size_t)HH * DD * sizeof(h16));
  h16* wkT = (h16*)alloc((size_t)HH * DD * sizeof(h16));
  h16* wvT = (h16*)alloc((size_t)HH * DD * sizeof(h16));
  h16* woT = (h16*)alloc((size_t)DD * HH * sizeof(h16));
  h16* qb  = (h16*)alloc(ML * HH * sizeof(h16));
  h16* kbf = (h16*)alloc(ML * HH * sizeof(h16));
  h16* vT  = (h16*)alloc((size_t)HH * ML * sizeof(h16));
  h16* at  = (h16*)alloc(ML * HH * sizeof(h16));
  (void)ws_size; (void)in_sizes; (void)n_in; (void)out_size;

  ca_cvt_f32_to_f16<<<8192, 256, 0, stream>>>(x, xb, (int)(ML * DD));
  ca_tcvt_f32_to_f16<<<1024, 256, 0, stream>>>(Wq, wqT, DD, HH);
  ca_tcvt_f32_to_f16<<<1024, 256, 0, stream>>>(Wk, wkT, DD, HH);
  ca_tcvt_f32_to_f16<<<1024, 256, 0, stream>>>(Wv, wvT, DD, HH);
  ca_tcvt_f32_to_f16<<<1024, 256, 0, stream>>>(Wo, woT, HH, DD);

  const dim3 blk(128);
  ca_gemm<<<dim3(1, 256), blk, 0, stream>>>(xb, wqT, bq, qb,
      (int)ML, HH, DD, DD, DD, HH, 0);
  ca_gemm<<<dim3(1, 256), blk, 0, stream>>>(xb, wkT, bk, kbf,
      (int)ML, HH, DD, DD, DD, HH, 0);
  ca_gemm<<<dim3(1, 256), blk, 0, stream>>>(xb, wvT, bv, vT,
      (int)ML, HH, DD, DD, DD, (int)ML, 1);  // store V transposed

  ca_flash_attn<<<256, blk, 0, stream>>>(qb, kbf, vT, at);

  ca_gemm<<<dim3(16, 256), blk, 0, stream>>>(at, woT, bo, (float*)d_out,
      (int)ML, DD, HH, HH, HH, DD, 2);
}
